// CoSSM_77206332113417
// MI455X (gfx1250) — matrix operand order repaired and checked by reading, not compiled, not run on hardware
//
#include <hip/hip_runtime.h>
#include <hip/hip_bf16.h>
#include <math.h>
#include <stdint.h>

#define DM   768
#define DS   16
#define DC   4
#define DI   1536
#define DTR  48
#define DTRP 64
#define BB   2
#define LL   2048
#define M4   (BB*LL)   /* 4096 rows for all GEMMs */
#define XD   80        /* DT_RANK + 2*D_STATE */

typedef __attribute__((ext_vector_type(16))) __bf16 v16bf;
typedef __attribute__((ext_vector_type(8)))  __bf16 v8bf;
typedef __attribute__((ext_vector_type(8)))  float  v8f;

#define CONCAT16(lo, hi) \
    __builtin_shufflevector(lo, hi, 0, 1, 2, 3, 4, 5, 6, 7, 8, 9, 10, 11, 12, 13, 14, 15)

// ---------------------------------------------------------------------------
// WMMA GEMM: C[M,N] = A[M,K] * W[N,K]^T  (A,W bf16 row-major).
// Fully specialized (K, strip, leading dims compile-time; grids exact) ->
// branch-free k-loop.  All loads for a k-step are issued first (single
// s_clause), then the NSTRIP WMMAs, so waits overlap with matrix issue.
//
//   A (16x32): lane = 16*half + m ; runs at k0+half*8 and k0+16+half*8
//   B (32x16): lane = 16*half + n ; runs at k0+half*8 and k0+16+half*8
//   C/D (16x16 f32): lane = 16*half + n ; VGPR v -> row m0 + half*8 + v
//
// MODE 0: store f32 row-major (LDC)
// MODE 1: store bf16 channel-major xz layout [b][n][l]   (in_proj)
// MODE 2: store f32 softplus(acc + bias[n])              (dtproj -> delta)
// ---------------------------------------------------------------------------
template <int MODE, int K, int NSTRIP, int LDA, int LDB, int LDC>
__global__ __launch_bounds__(128) void wmma_gemm(
        const __bf16* __restrict__ A, const __bf16* __restrict__ W,
        float* __restrict__ Cout, const float* __restrict__ bias,
        __bf16* __restrict__ xz_out) {
    const int lane = threadIdx.x & 31;
    const int wave = threadIdx.x >> 5;
    const int m0   = (blockIdx.x * 4 + wave) * 16;
    const int half = lane >> 4;
    const int mr   = lane & 15;          // row within A tile / col within B,C
    const int nbase = blockIdx.y * (NSTRIP * 16);

    const __bf16* ap = A + (size_t)(m0 + mr) * LDA + half * 8;
    const __bf16* bp = W + (size_t)(nbase + mr) * LDB + half * 8;

    v8f acc[NSTRIP];
#pragma unroll
    for (int j = 0; j < NSTRIP; ++j) acc[j] = v8f{};

#pragma unroll 2
    for (int k0 = 0; k0 < K; k0 += 32) {
        // speculative prefetch of next A chunk (unguarded: dropped if invalid)
        __builtin_prefetch(ap + k0 + 64, 0, 3);
        v8bf alo = *(const v8bf*)(ap + k0);
        v8bf ahi = *(const v8bf*)(ap + k0 + 16);
        v8bf bl[NSTRIP], bh[NSTRIP];
#pragma unroll
        for (int j = 0; j < NSTRIP; ++j) {
            const __bf16* bj = bp + (size_t)j * 16 * LDB + k0;
            bl[j] = *(const v8bf*)(bj);
            bh[j] = *(const v8bf*)(bj + 16);
        }
        v16bf a = CONCAT16(alo, ahi);
#pragma unroll
        for (int j = 0; j < NSTRIP; ++j) {
            v16bf b = CONCAT16(bl[j], bh[j]);
            acc[j] = __builtin_amdgcn_wmma_f32_16x16x32_bf16(
                false, a, false, b, (short)0, acc[j], false, false);
        }
    }

#pragma unroll
    for (int j = 0; j < NSTRIP; ++j) {
        const int n = nbase + j * 16 + mr;
#pragma unroll
        for (int v = 0; v < 8; ++v) {
            const int m = m0 + half * 8 + v;
            float val = acc[j][v];
            if (MODE == 0) {
                Cout[(size_t)m * LDC + n] = val;
            } else if (MODE == 1) {                // bf16, [b][channel][l]
                int b = m >> 11, l = m & (LL - 1);
                xz_out[((size_t)b * (2 * DI) + n) * LL + l] = (__bf16)val;
            } else {                               // softplus(dt + bias)
                float x = val + bias[n];
                Cout[(size_t)m * LDC + n] = (x > 20.f) ? x : log1pf(__expf(x));
            }
        }
    }
}

// ---------------------------------------------------------------------------
// Depthwise causal conv (width 4) + SiLU.  Input xz channel-major bf16,
// dir=1 reads the length-flipped sequence.  Output row-major [m][d] bf16
// so it feeds the xproj GEMM as the A matrix directly.
// ---------------------------------------------------------------------------
__global__ void conv_silu(const __bf16* __restrict__ xz, const float* __restrict__ w,
                          const float* __restrict__ bias, __bf16* __restrict__ xconv,
                          int dir) {
    int idx = blockIdx.x * blockDim.x + threadIdx.x;
    if (idx >= BB * LL * DI) return;
    int l = idx & (LL - 1);
    int rest = idx >> 11;
    int d = rest % DI;
    int b = rest / DI;
    const __bf16* xrow = xz + ((size_t)b * (2 * DI) + d) * LL;  // x half
    float s = bias[d];
#pragma unroll
    for (int j = 0; j < DC; ++j) {
        int t = l - (DC - 1) + j;
        if (t >= 0) {
            int li = dir ? (LL - 1 - t) : t;
            s += w[d * DC + j] * (float)xrow[li];
        }
    }
    float y = s / (1.f + __expf(-s));
    xconv[((size_t)b * LL + l) * DI + d] = (__bf16)y;
}

// Pad x_dbl[:, :48] -> bf16 [M4][64] (zero cols 48..63) for the dtproj GEMM.
__global__ void pad_xdbl(const float* __restrict__ xd, __bf16* __restrict__ out) {
    int i = blockIdx.x * blockDim.x + threadIdx.x;
    if (i >= M4 * DTRP) return;
    int r = i & 63, m = i >> 6;
    out[i] = (__bf16)((r < DTR) ? xd[(size_t)m * XD + r] : 0.f);
}

// Pad dtproj_w [DI][48] -> bf16 [DI][64].
__global__ void pad_dtw(const float* __restrict__ w, __bf16* __restrict__ out) {
    int i = blockIdx.x * blockDim.x + threadIdx.x;
    if (i >= DI * DTRP) return;
    int r = i & 63, n = i >> 6;
    out[i] = (__bf16)((r < DTR) ? w[n * DTR + r] : 0.f);
}

__global__ void f32_to_bf16(const float* __restrict__ in, __bf16* __restrict__ out, int n) {
    int i = blockIdx.x * blockDim.x + threadIdx.x;
    if (i < n) out[i] = (__bf16)in[i];
}

// ---------------------------------------------------------------------------
// Selective scan: one thread per channel d, 16 states in VGPRs.
// B/C vectors (32 floats/timestep, shared by all channels) are staged through
// LDS in 32-timestep chunks using the CDNA5 async global->LDS path
// (global_load_async_to_lds_b128, tracked by ASYNCcnt): 256 threads x 16B
// covers the 4KB chunk with one async op each.  Accumulates 0.5 * gated_y
// into out_acc at the (possibly flipped) time index; branch kernels run
// sequentially on the stream so plain += is race-free.
// ---------------------------------------------------------------------------
__global__ __launch_bounds__(256) void ssm_scan(
        const float* __restrict__ delta, const float* __restrict__ xdbl,
        const __bf16* __restrict__ xconv, const __bf16* __restrict__ xz,
        const float* __restrict__ A_log, const float* __restrict__ Dp,
        float* __restrict__ out_acc, int dir) {
    __shared__ float bc[32 * 32];
    const int tid = threadIdx.x;
    const int d = blockIdx.x * 256 + tid;
    const int b = blockIdx.y;
    float Ar[16], h[16];
#pragma unroll
    for (int n = 0; n < 16; ++n) { Ar[n] = -__expf(A_log[d * 16 + n]); h[n] = 0.f; }
    const float Dv = Dp[d];
    const __bf16* zrow = xz + ((size_t)b * (2 * DI) + DI + d) * LL;

    // Per-thread async-copy geometry: thread = (t within chunk, 16B segment).
    const int st  = tid >> 3;     // 0..31 : timestep within chunk
    const int seg = tid & 7;      // 0..7  : 16-byte segment of the 128B row
    const unsigned lds_off = (unsigned)(uintptr_t)&bc[st * 32 + seg * 4];

    for (int tc = 0; tc < LL / 32; ++tc) {
        {
            const float* src =
                xdbl + ((size_t)b * LL + tc * 32 + st) * XD + DTR + seg * 4;
            asm volatile("global_load_async_to_lds_b128 %0, %1, off"
                         :: "v"(lds_off), "v"((uint64_t)(uintptr_t)src)
                         : "memory");
            asm volatile("s_wait_asynccnt 0x0" ::: "memory");
        }
        __syncthreads();
        for (int tt = 0; tt < 32; ++tt) {
            int t = tc * 32 + tt;
            size_t m = (size_t)b * LL + t;
            float dlt = delta[m * DI + d];
            float xv  = (float)xconv[m * DI + d];
            float dx  = dlt * xv;
            float y   = 0.f;
#pragma unroll
            for (int n = 0; n < 16; ++n) {
                float e = __expf(dlt * Ar[n]);
                h[n] = e * h[n] + dx * bc[tt * 32 + n];
                y += h[n] * bc[tt * 32 + 16 + n];
            }
            y += xv * Dv;
            int lz = dir ? (LL - 1 - t) : t;
            float z = (float)zrow[lz];
            y *= z / (1.f + __expf(-z));
            out_acc[((size_t)b * LL + lz) * DI + d] += 0.5f * y;
        }
        __syncthreads();
    }
}

// ---------------------------------------------------------------------------
extern "C" void kernel_launch(void* const* d_in, const int* in_sizes, int n_in,
                              void* d_out, int out_size, void* d_ws, size_t ws_size,
                              hipStream_t stream) {
    (void)in_sizes; (void)n_in; (void)out_size; (void)ws_size;
    const float* g_hid = (const float*)d_in[0];
    const float* r_hid = (const float*)d_in[1];
    const float* g_inw = (const float*)d_in[2];
    const float* r_inw = (const float*)d_in[3];
    const float* A_log  = (const float*)d_in[24];
    const float* A_logb = (const float*)d_in[25];
    const float* g_outw = (const float*)d_in[30];
    const float* r_outw = (const float*)d_in[31];
    float* out = (float*)d_out;

    char* ws = (char*)d_ws;
    size_t off = 0;
    auto alloc = [&](size_t nbytes) -> void* {
        void* p = ws + off;
        off += (nbytes + 255) & ~(size_t)255;
        return p;
    };

    __bf16* hidg = (__bf16*)alloc((size_t)M4 * DM * 2);
    __bf16* hidr = (__bf16*)alloc((size_t)M4 * DM * 2);
    __bf16* wing = (__bf16*)alloc((size_t)2 * DI * DM * 2);
    __bf16* winr = (__bf16*)alloc((size_t)2 * DI * DM * 2);
    __bf16* woutg = (__bf16*)alloc((size_t)DM * DI * 2);
    __bf16* woutr = (__bf16*)alloc((size_t)DM * DI * 2);
    __bf16* wx[4];
    for (int i = 0; i < 4; ++i) wx[i] = (__bf16*)alloc((size_t)XD * DI * 2);
    __bf16* wdt[4];
    for (int i = 0; i < 4; ++i) wdt[i] = (__bf16*)alloc((size_t)DI * DTRP * 2);
    __bf16* xzg = (__bf16*)alloc((size_t)BB * 2 * DI * LL * 2);
    __bf16* xzr = (__bf16*)alloc((size_t)BB * 2 * DI * LL * 2);
    float* accg = (float*)alloc((size_t)M4 * DI * 4);
    float* accr = (float*)alloc((size_t)M4 * DI * 4);
    // branch-local scratch, reused (stream serializes branch kernels)
    __bf16* xconv = (__bf16*)alloc((size_t)M4 * DI * 2);
    float*  xdbl  = (float*)alloc((size_t)M4 * XD * 4);
    __bf16* dtin  = (__bf16*)alloc((size_t)M4 * DTRP * 2);
    float*  dlt   = (float*)alloc((size_t)M4 * DI * 4);
    __bf16* accb  = (__bf16*)alloc((size_t)M4 * DI * 2);

    auto cvt = [&](const float* src, __bf16* dst, int n) {
        f32_to_bf16<<<(n + 255) / 256, 256, 0, stream>>>(src, dst, n);
    };
    cvt(g_hid, hidg, M4 * DM);
    cvt(r_hid, hidr, M4 * DM);
    cvt(g_inw, wing, 2 * DI * DM);
    cvt(r_inw, winr, 2 * DI * DM);
    cvt(g_outw, woutg, DM * DI);
    cvt(r_outw, woutr, DM * DI);
    const int xw_idx[4] = {12, 14, 13, 15};        // gf, gb, rf, rb
    const int dw_idx[4] = {16, 20, 18, 22};
    for (int i = 0; i < 4; ++i) {
        cvt((const float*)d_in[xw_idx[i]], wx[i], XD * DI);
        pad_dtw<<<(DI * DTRP + 255) / 256, 256, 0, stream>>>(
            (const float*)d_in[dw_idx[i]], wdt[i]);
    }

    // in_proj GEMMs: M=4096, N=3072, K=768 -> xz (bf16, channel-major)
    {
        dim3 g(M4 / 64, (2 * DI) / 64);
        wmma_gemm<1, DM, 4, DM, DM, 0><<<g, 128, 0, stream>>>(
            hidg, wing, nullptr, nullptr, xzg);
        wmma_gemm<1, DM, 4, DM, DM, 0><<<g, 128, 0, stream>>>(
            hidr, winr, nullptr, nullptr, xzr);
    }

    hipMemsetAsync(accg, 0, (size_t)M4 * DI * 4, stream);
    hipMemsetAsync(accr, 0, (size_t)M4 * DI * 4, stream);

    // branches: {g fwd, g bwd, r fwd, r bwd}
    const int cw_idx[4] = {4, 8, 6, 10};
    const int cb_idx[4] = {5, 9, 7, 11};
    const int db_idx[4] = {17, 21, 19, 23};
    const int dD_idx[4] = {26, 28, 27, 29};
    for (int br = 0; br < 4; ++br) {
        int dir = br & 1;
        const __bf16* xz = (br < 2) ? xzg : xzr;
        float* acc = (br < 2) ? accg : accr;
        const float* Al = dir ? A_logb : A_log;

        conv_silu<<<(BB * LL * DI + 255) / 256, 256, 0, stream>>>(
            xz, (const float*)d_in[cw_idx[br]], (const float*)d_in[cb_idx[br]],
            xconv, dir);

        // xproj: M=4096, N=80 (5 tiles, one strip), K=1536
        dim3 gx(M4 / 64, 1);
        wmma_gemm<0, DI, 5, DI, DI, XD><<<gx, 128, 0, stream>>>(
            xconv, wx[br], xdbl, nullptr, nullptr);

        pad_xdbl<<<(M4 * DTRP + 255) / 256, 256, 0, stream>>>(xdbl, dtin);

        // dtproj: M=4096, N=1536, K=64 (padded), softplus+bias epilogue
        dim3 gd(M4 / 64, DI / 64);
        wmma_gemm<2, DTRP, 4, DTRP, DTRP, DI><<<gd, 128, 0, stream>>>(
            dtin, wdt[br], dlt, (const float*)d_in[db_idx[br]], nullptr);

        dim3 gs(DI / 256, BB);
        ssm_scan<<<gs, 256, 0, stream>>>(dlt, xdbl, xconv, xz, Al,
                                         (const float*)d_in[dD_idx[br]], acc, dir);
    }

    // output GEMMs: M=4096, N=768, K=1536 -> d_out (g then r)
    {
        dim3 go(M4 / 64, DM / 64);
        cvt(accg, accb, M4 * DI);
        wmma_gemm<0, DI, 4, DI, DI, DM><<<go, 128, 0, stream>>>(
            accb, woutg, out, nullptr, nullptr);
        cvt(accr, accb, M4 * DI);
        wmma_gemm<0, DI, 4, DI, DI, DM><<<go, 128, 0, stream>>>(
            accb, woutr, out + (size_t)M4 * DM, nullptr, nullptr);
    }
}
